// VarAttention_24584392803108
// MI455X (gfx1250) — matrix operand order repaired and checked by reading, not compiled, not run on hardware
//
#include <hip/hip_runtime.h>
#include <hip/hip_bf16.h>

// ---------------------------------------------------------------------------
// Types for CDNA5 WMMA (wave32): v16bf A/B fragments, v8f accumulator.
// ---------------------------------------------------------------------------
typedef __bf16 bf16;
typedef __attribute__((__ext_vector_type__(4)))  bf16  bf16x4;
typedef __attribute__((__ext_vector_type__(8)))  bf16  bf16x8;
typedef __attribute__((__ext_vector_type__(16))) bf16  bf16x16;
typedef __attribute__((__ext_vector_type__(8)))  float floatx8;
typedef int v4i_vs __attribute__((vector_size(16)));   // 16B chunk for async copy

#define SHUF16(lo, hi) __builtin_shufflevector((lo), (hi), 0,1,2,3,4,5,6,7,8,9,10,11,12,13,14,15)

#if __has_builtin(__builtin_amdgcn_global_load_async_to_lds_b128) && \
    __has_builtin(__builtin_amdgcn_s_wait_asynccnt)
#define HAVE_ASYNC 1
#else
#define HAVE_ASYNC 0
#endif

__device__ __forceinline__ bf16 f2bf(float f) {
  unsigned u = __builtin_bit_cast(unsigned, f);
  u += 0x7FFFu + ((u >> 16) & 1u);          // round-to-nearest-even
  unsigned short s = (unsigned short)(u >> 16);
  return __builtin_bit_cast(bf16, s);
}

// 16B global -> LDS copy; async (ASYNCcnt-tracked) when available.
__device__ __forceinline__ void copy16_g2l(const bf16* g, bf16* l) {
#if HAVE_ASYNC
  __builtin_amdgcn_global_load_async_to_lds_b128(
      (__attribute__((address_space(1))) v4i_vs*)g,
      (__attribute__((address_space(3))) v4i_vs*)l, 0, 0);
#else
  *(bf16x8*)l = *(const bf16x8*)g;
#endif
}

__device__ __forceinline__ floatx8 wmma_bf16(bf16x16 a, bf16x16 b, floatx8 c) {
  return __builtin_amdgcn_wmma_f32_16x16x32_bf16(false, a, false, b, (short)0, c, false, false);
}

// Problem constants
#define BB 8
#define NN 512
#define PP 16
#define CC 512
#define HH 8
#define HD 64

// ---------------------------------------------------------------------------
// Kernel: bulk fp32 -> bf16 conversion (8 elements / thread).
// ---------------------------------------------------------------------------
__global__ __launch_bounds__(256) void convert_kernel(
    const float* __restrict__ src, bf16* __restrict__ dst) {
  size_t i = (size_t)blockIdx.x * 256 + threadIdx.x;   // chunk of 8
  float4 a = ((const float4*)src)[i * 2];
  float4 b = ((const float4*)src)[i * 2 + 1];
  bf16x8 o = { f2bf(a.x), f2bf(a.y), f2bf(a.z), f2bf(a.w),
               f2bf(b.x), f2bf(b.y), f2bf(b.z), f2bf(b.w) };
  ((bf16x8*)dst)[i] = o;
}

// ---------------------------------------------------------------------------
// Kernel: weight transpose + bf16 cast: src[K x N] fp32 -> dst[N x K] bf16.
// ---------------------------------------------------------------------------
__global__ __launch_bounds__(256) void wtrans_kernel(
    const float* __restrict__ src, bf16* __restrict__ dst, int K, int Ncols) {
  int idx = blockIdx.x * 256 + threadIdx.x;   // over src, coalesced reads
  int k = idx / Ncols, n = idx % Ncols;
  dst[(size_t)n * K + k] = f2bf(src[idx]);
}

// ---------------------------------------------------------------------------
// Kernel: masked mean of x over P (from bf16 x) -> xm bf16 [B*N, C]
// k = mean_p(mask * (x W_k)) = (mean_p(mask * x)) W_k  (linearity)
// ---------------------------------------------------------------------------
__global__ __launch_bounds__(256) void maskmean_kernel(
    const bf16* __restrict__ xbf, const float* __restrict__ mask,
    bf16* __restrict__ xm) {
  int idx = blockIdx.x * 256 + threadIdx.x;   // 0 .. B*N*C-1
  int c  = idx & (CC - 1);
  int bn = idx >> 9;
  float s = 0.f, ms = 0.f;
#pragma unroll
  for (int p = 0; p < PP; ++p) {
    float mv = mask[bn * PP + p];
    s  += mv * (float)xbf[((size_t)(bn * PP + p)) * CC + c];
    ms += mv;
  }
  xm[idx] = f2bf(s / ms);
}

// ---------------------------------------------------------------------------
// Kernel: bf16 TN GEMM.  D = A[M x K] (row stride lda) * Bt[Ncols x K]^T
// Double-buffered LDS: async copies for tile t+1 run behind tile t's WMMAs
// (async loads complete in order -> s_wait_asynccnt 3 isolates tile t).
// Output: bf16 (Dbf) or fp32+bias (Df).
// BM=128, BN=64, BK=32; 8 waves, each a 32x32 subtile (2x2 WMMA tiles).
// Requires M%128==0, Ncols%64==0, K%32==0.
// ---------------------------------------------------------------------------
#define BM 128
#define BN 64
#define BK 32
#define LPITCH 40   // halfwords; 80B rows keep 16B alignment, skew banks

__global__ __launch_bounds__(256) void gemm_tn(
    const bf16* __restrict__ A, int lda,
    const bf16* __restrict__ Bt, int ldbt, int K,
    bf16* __restrict__ Dbf, float* __restrict__ Df, int ldc,
    const float* __restrict__ bias) {
  __shared__ bf16 As[2][BM * LPITCH];   // double-buffered [128][32]
  __shared__ bf16 Bs[2][BN * LPITCH];   // double-buffered [64][32]

  int tid  = threadIdx.x;
  int wave = tid >> 5, lane = tid & 31;
  int waveM = wave >> 1, waveN = wave & 1;
  int l15 = lane & 15;
  int hi  = (lane >= 16);
  int m0 = blockIdx.y * BM;
  int n0 = blockIdx.x * BN;

  // staging coordinates (16B chunks of 8 halfwords)
  int arow0 = tid >> 2, ac8 = (tid & 3) << 3;          // A chunk 0
  int arow1 = (tid + 256) >> 2;                        // A chunk 1
  int brow  = tid >> 2, bc8 = (tid & 3) << 3;          // B chunk

  const bf16* a0 = A  + (size_t)(m0 + arow0) * lda  + ac8;
  const bf16* a1 = A  + (size_t)(m0 + arow1) * lda  + ac8;
  const bf16* b0 = Bt + (size_t)(n0 + brow)  * ldbt + bc8;

  floatx8 acc[2][2] = {};
  int nk = K / BK;

  // prologue: stage tile 0 into buffer 0
  copy16_g2l(a0, &As[0][arow0 * LPITCH + ac8]);
  copy16_g2l(a1, &As[0][arow1 * LPITCH + ac8]);
  copy16_g2l(b0, &Bs[0][brow  * LPITCH + bc8]);

  for (int t = 0; t < nk; ++t) {
    int cur = t & 1, nxt = cur ^ 1;
    if (t + 1 < nk) {
      int kn = (t + 1) * BK;
      copy16_g2l(a0 + kn, &As[nxt][arow0 * LPITCH + ac8]);
      copy16_g2l(a1 + kn, &As[nxt][arow1 * LPITCH + ac8]);
      copy16_g2l(b0 + kn, &Bs[nxt][brow  * LPITCH + bc8]);
      if (t + 2 < nk) {   // WGP-scope prefetch two tiles ahead
        __builtin_prefetch(a0 + kn + BK, 0, 3);
        __builtin_prefetch(b0 + kn + BK, 0, 3);
      }
#if HAVE_ASYNC
      __builtin_amdgcn_s_wait_asynccnt(3);   // tile t's 3 copies done
#endif
    } else {
#if HAVE_ASYNC
      __builtin_amdgcn_s_wait_asynccnt(0);
#endif
    }
    __syncthreads();     // tile t visible to all waves

    const bf16* Asb = As[cur];
    const bf16* Bsb = Bs[cur];
    int kA = hi ? 8 : 0;     // A frag: lanes<16 hold K 0..7 & 16..23
    int kB = hi ? 8 : 0;     // B frag: lanes<16 hold K 0..7 & 16..23
    bf16x16 afrag[2], bfrag[2];
#pragma unroll
    for (int tm = 0; tm < 2; ++tm) {
      int row = waveM * 32 + tm * 16 + l15;
      bf16x8 lo = *(const bf16x8*)(&Asb[row * LPITCH + kA]);
      bf16x8 h8 = *(const bf16x8*)(&Asb[row * LPITCH + kA + 16]);
      afrag[tm] = SHUF16(lo, h8);
    }
#pragma unroll
    for (int tn = 0; tn < 2; ++tn) {
      int col = waveN * 32 + tn * 16 + l15;
      bf16x8 lo = *(const bf16x8*)(&Bsb[col * LPITCH + kB]);
      bf16x8 h8 = *(const bf16x8*)(&Bsb[col * LPITCH + kB + 16]);
      bfrag[tn] = SHUF16(lo, h8);
    }
#pragma unroll
    for (int tm = 0; tm < 2; ++tm)
#pragma unroll
      for (int tn = 0; tn < 2; ++tn)
        acc[tm][tn] = wmma_bf16(afrag[tm], bfrag[tn], acc[tm][tn]);
    __syncthreads();     // done reading buffer `cur`; reusable at t+2
  }

  int mo = hi ? 8 : 0;
#pragma unroll
  for (int tm = 0; tm < 2; ++tm)
#pragma unroll
    for (int tn = 0; tn < 2; ++tn) {
      int col = n0 + waveN * 32 + tn * 16 + l15;
      if (Dbf) {
#pragma unroll
        for (int r = 0; r < 8; ++r) {
          int m = m0 + waveM * 32 + tm * 16 + r + mo;
          Dbf[(size_t)m * ldc + col] = f2bf(acc[tm][tn][r]);
        }
      } else {
        float bv = bias ? bias[col] : 0.f;
#pragma unroll
        for (int r = 0; r < 8; ++r) {
          int m = m0 + waveM * 32 + tm * 16 + r + mo;
          Df[(size_t)m * ldc + col] = acc[tm][tn][r] + bv;
        }
      }
    }
}

// ---------------------------------------------------------------------------
// Kernel: fused SDPA per (b, h, 32-row tile).  All inputs bf16.
//   q:[B*N, C] head slice, k:[B*N, C], v:[B*N*P, C] native layout.
//   out -> aout bf16 in [b,n,p,c] layout, ready for proj GEMM.
// ---------------------------------------------------------------------------
__global__ __launch_bounds__(256) void attn_fused(
    const bf16* __restrict__ qbuf, const bf16* __restrict__ kbuf,
    const bf16* __restrict__ vbuf, bf16* __restrict__ aout) {
  __shared__ bf16  qs[32 * 72];        // q tile [32][64]
  __shared__ float S[32 * 512];        // fp32 scores
  __shared__ bf16  Pbf[32 * 520];      // softmax probs
  __shared__ bf16  VT[256 * 40];       // V tile transposed [e][k]
  __shared__ float red[288];

  int b = blockIdx.z, h = blockIdx.y, mb = blockIdx.x;
  int tid = threadIdx.x, wave = tid >> 5, lane = tid & 31;
  int l15 = lane & 15;
  int hi  = (lane >= 16);
  int hi8 = hi ? 8 : 0;
  int m0 = mb * 32;

  // ---------------- phase 1: S = (q k^T) * 1/sqrt(64) ----------------
  {  // stage q tile 32x64: 256 chunks of 8 halfwords (async if available)
    int row = tid >> 3, c8 = (tid & 7) << 3;
    copy16_g2l(qbuf + (size_t)(b * NN + m0 + row) * CC + h * HD + c8,
               &qs[row * 72 + c8]);
#if HAVE_ASYNC
    __builtin_amdgcn_s_wait_asynccnt(0);
#endif
  }
  __syncthreads();

  const float scale = 0.125f;
  for (int j = 0; j < 4; ++j) {        // each wave owns score cols [wave*64, +64)
    int ncol0 = wave * 64 + j * 16;
    floatx8 acc[2] = {};
#pragma unroll
    for (int ks = 0; ks < 2; ++ks) {
      // B frag = k^T: lane's column is key index m; d contiguous in global
      int mcol = ncol0 + l15;
      int d0 = ks * 32 + (hi ? 8 : 0);
      const bf16* kp = kbuf + (size_t)(b * NN + mcol) * CC + h * HD + d0;
      bf16x8 klo = *(const bf16x8*)(kp);
      bf16x8 khi = *(const bf16x8*)(kp + 16);
      bf16x16 bb = SHUF16(klo, khi);

      int kA = ks * 32 + (hi ? 8 : 0);
#pragma unroll
      for (int mt = 0; mt < 2; ++mt) {
        int row = mt * 16 + l15;
        bf16x8 lo = *(const bf16x8*)(&qs[row * 72 + kA]);
        bf16x8 hh = *(const bf16x8*)(&qs[row * 72 + kA + 16]);
        bf16x16 a = SHUF16(lo, hh);
        acc[mt] = wmma_bf16(a, bb, acc[mt]);
      }
    }
#pragma unroll
    for (int mt = 0; mt < 2; ++mt)
#pragma unroll
      for (int r = 0; r < 8; ++r) {
        int ml = mt * 16 + r + hi8;
        S[ml * 512 + ncol0 + l15] = acc[mt][r] * scale;
      }
  }
  __syncthreads();

  // ---------------- phase 2: row softmax over 512 -> bf16 Pbf ----------------
  {
    int srow = tid & 31, seg = tid >> 5;   // 8 segments x 64 cols
    int c0 = seg * 64;
    float mx = -1e30f;
    for (int c = 0; c < 64; ++c) mx = fmaxf(mx, S[srow * 512 + c0 + c]);
    red[seg * 32 + srow] = mx;
    __syncthreads();
    if (tid < 32) {
      float m = -1e30f;
#pragma unroll
      for (int s = 0; s < 8; ++s) m = fmaxf(m, red[s * 32 + tid]);
      red[256 + tid] = m;
    }
    __syncthreads();
    float rm = red[256 + srow];
    float sum = 0.f;
    for (int c = 0; c < 64; ++c) {
      float e = __expf(S[srow * 512 + c0 + c] - rm);
      S[srow * 512 + c0 + c] = e;
      sum += e;
    }
    red[seg * 32 + srow] = sum;
    __syncthreads();
    if (tid < 32) {
      float s = 0.f;
#pragma unroll
      for (int q = 0; q < 8; ++q) s += red[q * 32 + tid];
      red[256 + tid] = 1.0f / s;
    }
    __syncthreads();
    float inv = red[256 + srow];
    for (int c = 0; c < 64; ++c)
      Pbf[srow * 520 + c0 + c] = f2bf(S[srow * 512 + c0 + c] * inv);
    __syncthreads();
  }

  // ---------------- phase 3: O = P @ V (packed e = hd*16 + p) ----------------
  const bf16* vbase = vbuf + (size_t)b * NN * (PP * CC) + h * HD;
  for (int eb = 0; eb < 4; ++eb) {       // e blocks of 256 (hd blocks of 16)
    floatx8 acc[2][2] = {};
    for (int ks = 0; ks < 16; ++ks) {    // K = 512 keys, steps of 32
      __syncthreads();                   // protect VT from previous iteration
      // stage VT[e_local][k]; hd contiguous in global (4 bf16 = 8B loads)
#pragma unroll
      for (int i = 0; i < 8; ++i) {
        int f = tid + i * 256;
        int k = f >> 6;
        int rem = f & 63;
        int p = rem >> 2, hd4 = rem & 3;
        bf16x4 v = *(const bf16x4*)(vbase + (size_t)(ks * 32 + k) * (PP * CC) +
                                    p * CC + eb * 16 + hd4 * 4);
#pragma unroll
        for (int jj = 0; jj < 4; ++jj) {
          int el = (hd4 * 4 + jj) * 16 + p;
          VT[el * 40 + k] = v[jj];
        }
      }
      __syncthreads();

      int kB = hi ? 8 : 0;
      int kA = ks * 32 + (hi ? 8 : 0);
#pragma unroll
      for (int et = 0; et < 2; ++et) {
        int el = wave * 32 + et * 16 + l15;
        bf16x8 lo = *(const bf16x8*)(&VT[el * 40 + kB]);
        bf16x8 hh = *(const bf16x8*)(&VT[el * 40 + kB + 16]);
        bf16x16 bbv = SHUF16(lo, hh);
#pragma unroll
        for (int mt = 0; mt < 2; ++mt) {
          int prow = mt * 16 + l15;
          bf16x8 alo = *(const bf16x8*)(&Pbf[prow * 520 + kA]);
          bf16x8 ahi = *(const bf16x8*)(&Pbf[prow * 520 + kA + 16]);
          bf16x16 a = SHUF16(alo, ahi);
          acc[et][mt] = wmma_bf16(a, bbv, acc[et][mt]);
        }
      }
    }
    // store O into [b,n,p,c] layout (un-packs P for free)
#pragma unroll
    for (int et = 0; et < 2; ++et)
#pragma unroll
      for (int mt = 0; mt < 2; ++mt)
#pragma unroll
        for (int r = 0; r < 8; ++r) {
          int nrow = m0 + mt * 16 + r + hi8;
          int e = eb * 256 + wave * 32 + et * 16 + l15;
          int p = e & 15, hd = e >> 4;
          aout[((size_t)((b * NN + nrow) * PP + p)) * CC + h * HD + hd] =
              f2bf(acc[et][mt][r]);
        }
  }
}

// ---------------------------------------------------------------------------
// Launch
// ---------------------------------------------------------------------------
extern "C" void kernel_launch(void* const* d_in, const int* in_sizes, int n_in,
                              void* d_out, int out_size, void* d_ws, size_t ws_size,
                              hipStream_t stream) {
  (void)in_sizes; (void)n_in; (void)out_size; (void)ws_size;
  const float* x     = (const float*)d_in[0];
  const float* mask  = (const float*)d_in[1];
  // d_in[2]=lens, d_in[3]=lens_mask: unused by reference forward
  const float* Wqkv  = (const float*)d_in[4];   // [512, 1536] row-major
  const float* Wproj = (const float*)d_in[5];   // [512, 512]
  const float* bproj = (const float*)d_in[6];   // [512]
  float* out = (float*)d_out;

  bf16* wsb   = (bf16*)d_ws;
  bf16* xbf   = wsb;                      // 33,554,432  x as bf16
  bf16* vb    = wsb + 33554432ull;        // 33,554,432  v = x @ Wv
  bf16* ao    = wsb + 67108864ull;        // 33,554,432  attention out [b,n,p,c]
  bf16* xmbf  = wsb + 100663296ull;       //  2,097,152  masked mean x
  bf16* qb    = wsb + 102760448ull;       //  2,097,152  q (p=0 rows)
  bf16* kb    = wsb + 104857600ull;       //  2,097,152  k
  bf16* wqkvT = wsb + 106954752ull;       //    786,432  Wqkv^T bf16 [1536][512]
  bf16* wprjT = wsb + 107741184ull;       //    262,144  Wproj^T bf16 [512][512]

  // pre-convert / pre-transpose (memory-bound, one pass)
  convert_kernel<<<dim3(16384), 256, 0, stream>>>(x, xbf);
  wtrans_kernel<<<dim3(3072), 256, 0, stream>>>(Wqkv, wqkvT, CC, 3 * CC);
  wtrans_kernel<<<dim3(1024), 256, 0, stream>>>(Wproj, wprjT, CC, CC);
  maskmean_kernel<<<dim3((BB * NN * CC) / 256), 256, 0, stream>>>(xbf, mask, xmbf);

  // q: only p==0 rows of x (row stride P*C), Wqkv^T rows 0..511
  gemm_tn<<<dim3(CC / BN, (BB * NN) / BM), 256, 0, stream>>>(
      xbf, PP * CC, wqkvT, CC, CC, qb, nullptr, CC, nullptr);
  // k: masked-mean x @ Wqkv[:, C:2C]   (Wqkv^T rows 512..1023)
  gemm_tn<<<dim3(CC / BN, (BB * NN) / BM), 256, 0, stream>>>(
      xmbf, CC, wqkvT + (size_t)CC * CC, CC, CC, kb, nullptr, CC, nullptr);
  // v: all rows @ Wqkv[:, 2C:3C]       (Wqkv^T rows 1024..1535)
  gemm_tn<<<dim3(CC / BN, (BB * NN * PP) / BM), 256, 0, stream>>>(
      xbf, CC, wqkvT + (size_t)2 * CC * CC, CC, CC, vb, nullptr, CC, nullptr);
  // fused attention
  attn_fused<<<dim3(NN / 32, HH, BB), 256, 0, stream>>>(qb, kb, vb, ao);
  // projection + bias -> fp32 output
  gemm_tn<<<dim3(CC / BN, (BB * NN * PP) / BM), 256, 0, stream>>>(
      ao, CC, wprjT, CC, CC, nullptr, out, CC, bproj);
}
